// audioSTRFAE_44478681318191
// MI455X (gfx1250) — matrix operand order repaired and checked by reading, not compiled, not run on hardware
//
#include <hip/hip_runtime.h>
#include <hip/hip_bf16.h>
#include <math.h>

typedef __attribute__((ext_vector_type(16))) _Float16 v16h;
typedef __attribute__((ext_vector_type(8)))  _Float16 v8h;
typedef __attribute__((ext_vector_type(8)))  float    v8f;

#define BATCH   16
#define T_IN    16000
#define N_CH    129
#define NFFT    256
#define HOP     80
#define NFRAMES 200
#define NROWS   (BATCH*NFRAMES)      /* 3200 GEMM rows           */
#define NPAD    144                  /* 129 freq bins padded     */
#define KSPEC   272                  /* 258 re/im padded, mult of 16 */
#define XT      13                   /* ceil(200/16) time tiles  */
#define PI_F    3.14159265358979f
#define ALPHA_F 0.99221790800f       /* exp(-1/128) */

// ---------------------------------------------------------------------------
// WMMA helpers (v_wmma_f32_16x16x32_f16, wave32, ISA 7.12.2 layouts)
// ---------------------------------------------------------------------------
__device__ __forceinline__ v16h combine16(v8h lo, v8h hi) {
  v16h r;
#pragma unroll
  for (int i = 0; i < 8; ++i) { r[i] = lo[i]; r[i + 8] = hi[i]; }
  return r;
}
// A row-major (M x K): lane<16 -> K kk+[0..7] & kk+[16..23]; lane>=16 -> +8
__device__ __forceinline__ v16h ldA16(const _Float16* __restrict__ row, int lane) {
  const int ko = (lane & 16) ? 8 : 0;
  v8h lo = *(const v8h*)(row + ko);
  v8h hi = *(const v8h*)(row + ko + 16);
  return combine16(lo, hi);
}
// B stored transposed (N x K) row-major: lane<16 -> K kk+[0..7] & kk+[16..23]; lane>=16 -> +8
__device__ __forceinline__ v16h ldB16(const _Float16* __restrict__ row, int lane) {
  const int ko = (lane & 16) ? 8 : 0;
  v8h lo = *(const v8h*)(row + ko);
  v8h hi = *(const v8h*)(row + ko + 16);
  return combine16(lo, hi);
}
__device__ __forceinline__ v8f wmma_f16(v16h a, v16h b, v8f c) {
  return __builtin_amdgcn_wmma_f32_16x16x32_f16(false, a, false, b, (short)0, c,
                                                false, false);
}

__device__ __forceinline__ float geluf(float x) {
  float x3 = x * x * x;
  return 0.5f * x * (1.0f + tanhf(0.7978845608f * (x + 0.044715f * x3)));
}
__device__ __forceinline__ float sigmoidf(float x) { return 1.0f / (1.0f + expf(-x)); }
__device__ __forceinline__ float hannf(int n) {
  return 0.5f - 0.5f * cosf((2.0f * PI_F * (float)n) / (float)NFFT);
}

// ---------------------------------------------------------------------------
// Table builders
// ---------------------------------------------------------------------------
__global__ void k_build_dft(_Float16* __restrict__ Bt) {
  int j = blockIdx.x, n = threadIdx.x;
  float v = 0.0f;
  float w = (2.0f * PI_F * (float)n) / (float)NFFT;
  if (j < NPAD) { if (j < N_CH) v = cosf(w * (float)j); }
  else { int k = j - NPAD; if (k < N_CH) v = -sinf(w * (float)k); }
  Bt[(size_t)j * NFFT + n] = (_Float16)v;
}
__global__ void k_build_idft(_Float16* __restrict__ Bt2) {
  int n = blockIdx.x, k = threadIdx.x;
  float wn = hannf(n);
  float v = 0.0f;
  if (k < N_CH) {
    float c = (k == 0 || k == 128) ? 1.0f : 2.0f;
    v = (c / (float)NFFT) * cosf((2.0f * PI_F * (float)k * (float)n) / (float)NFFT) * wn;
  } else if (k >= 136 && k < 136 + N_CH) {
    int kq = k - 136;
    float c = (kq == 0 || kq == 128) ? 1.0f : 2.0f;
    v = -(c / (float)NFFT) * sinf((2.0f * PI_F * (float)kq * (float)n) / (float)NFFT) * wn;
  }
  Bt2[(size_t)n * KSPEC + k] = (_Float16)v;
}
__global__ void k_build_weights(const float* __restrict__ K1,
                                const float* __restrict__ Wd,
                                _Float16* __restrict__ Wt2,
                                _Float16* __restrict__ Wdt) {
  int idx = blockIdx.x * blockDim.x + threadIdx.x;
  if (idx < 9 * 32 * 32) {
    int tap = idx >> 10, r = idx & 1023, oc = r >> 5, ic = r & 31;
    Wt2[idx] = (_Float16)K1[tap * 1024 + ic * 32 + oc];
  } else {
    int j = idx - 9 * 32 * 32;
    if (j < NPAD * 128) {
      int n = j >> 7, k = j & 127;
      Wdt[j] = (n < N_CH) ? (_Float16)Wd[k * N_CH + n] : (_Float16)0.0f;
    }
  }
}
__global__ void k_build_denom(float* __restrict__ denom, int OT) {
  int s = blockIdx.x * blockDim.x + threadIdx.x;
  if (s >= OT) return;
  float acc = 0.0f;
  for (int t = 0; t < NFRAMES; ++t) {
    int n = s + NFFT / 2 - t * HOP;
    if (n >= 0 && n < NFFT) { float w = hannf(n); acc += w * w; }
  }
  denom[s] = fmaxf(acc, 1e-6f);
}

// ---------------------------------------------------------------------------
// Windowed frame matrix  F[3200][256] (f16)
// ---------------------------------------------------------------------------
__global__ void k_frames(const float* __restrict__ x, _Float16* __restrict__ F) {
  int idx = blockIdx.x * blockDim.x + threadIdx.x;
  if (idx >= NROWS * NFFT) return;
  int row = idx >> 8, n = idx & 255;
  int b = row / NFRAMES, t = row % NFRAMES;
  int s = t * HOP + n - NFFT / 2;
  float v = (s >= 0 && s < T_IN) ? x[b * T_IN + s] * hannf(n) : 0.0f;
  F[idx] = (_Float16)v;
}

// ---------------------------------------------------------------------------
// Generic f16 GEMM, 8 waves per block, one 16x16 tile per wave.
// C[M][N] = A[M][K] * Bt[N][K]^T   grid.x = Mtiles*Ntiles/8
// ---------------------------------------------------------------------------
__global__ void k_gemm_f16(const _Float16* __restrict__ A,
                           const _Float16* __restrict__ Bt,
                           float* __restrict__ C, int K, int N,
                           int Mtiles, int Ntiles) {
  const int lane = threadIdx.x & 31;
  const int tid = blockIdx.x * 8 + (threadIdx.x >> 5);
  if (tid >= Mtiles * Ntiles) return;
  const int m0 = (tid % Mtiles) * 16, n0 = (tid / Mtiles) * 16;
  const _Float16* arow = A + (size_t)(m0 + (lane & 15)) * K;
  const _Float16* brow = Bt + (size_t)(n0 + (lane & 15)) * K;
  v8f acc = {};
  for (int kk = 0; kk < K; kk += 32) {
    __builtin_prefetch(arow + kk + 128, 0, 0);
    v16h a = ldA16(arow + kk, lane);
    v16h b = ldB16(brow + kk, lane);
    acc = wmma_f16(a, b, acc);
  }
  const int n = n0 + (lane & 15);
  const int mb = m0 + ((lane & 16) ? 8 : 0);
#pragma unroll
  for (int r = 0; r < 8; ++r) C[(size_t)(mb + r) * N + n] = acc[r];
}

// ---------------------------------------------------------------------------
// Cochlear IIR scan -> auditory spectrogram aud[B][128][200]
// ---------------------------------------------------------------------------
__global__ void k_iir_aud(const float* __restrict__ x, const float* __restrict__ Bs,
                          const float* __restrict__ As, float* __restrict__ aud) {
  const int b = blockIdx.x, t = threadIdx.x;   // t in [0,128)
  __shared__ float xs[400];
  float bc[2][5], ac[2][4], z[2][4];
  float w = 0.0f;
#pragma unroll
  for (int c = 0; c < 2; ++c) {
    int ch = t + c;
#pragma unroll
    for (int j = 0; j < 5; ++j) bc[c][j] = Bs[ch * 5 + j];
#pragma unroll
    for (int j = 0; j < 4; ++j) { ac[c][j] = As[ch * 5 + 1 + j]; z[c][j] = 0.0f; }
  }
  for (int base = 0; base < T_IN; base += 400) {
    for (int i = t; i < 400; i += 128) xs[i] = x[b * T_IN + base + i];
    __syncthreads();
    for (int i = 0; i < 400; ++i) {
      float xv = xs[i];
      float y[2];
#pragma unroll
      for (int c = 0; c < 2; ++c) {
        float yv = fmaf(bc[c][0], xv, z[c][0]);
        z[c][0] = fmaf(bc[c][1], xv, z[c][1]) - ac[c][0] * yv;
        z[c][1] = fmaf(bc[c][2], xv, z[c][2]) - ac[c][1] * yv;
        z[c][2] = fmaf(bc[c][3], xv, z[c][3]) - ac[c][2] * yv;
        z[c][3] = bc[c][4] * xv - ac[c][3] * yv;
        y[c] = yv;
      }
      w = fmaf(ALPHA_F, w, fmaxf(y[0] - y[1], 0.0f));
      int g = base + i;
      if ((g % HOP) == HOP - 1)
        aud[((size_t)b * 128 + t) * NFRAMES + g / HOP] = w;
    }
    __syncthreads();
  }
}

// ---------------------------------------------------------------------------
// conv1: 1->32 direct, GELU, f16 NHWC out
// ---------------------------------------------------------------------------
__global__ void k_conv1(const float* __restrict__ aud, const float* __restrict__ K0,
                        const float* __restrict__ c0, _Float16* __restrict__ h1) {
  int idx = blockIdx.x * blockDim.x + threadIdx.x;
  if (idx >= BATCH * 128 * NFRAMES) return;
  int xp = idx % NFRAMES, y = (idx / NFRAMES) % 128, b = idx / (128 * NFRAMES);
  float in[9];
#pragma unroll
  for (int dy = 0; dy < 3; ++dy)
#pragma unroll
    for (int dx = 0; dx < 3; ++dx) {
      int yy = y + dy - 1, xx = xp + dx - 1;
      in[dy * 3 + dx] = (yy >= 0 && yy < 128 && xx >= 0 && xx < NFRAMES)
                            ? aud[((size_t)b * 128 + yy) * NFRAMES + xx] : 0.0f;
    }
  _Float16* o = h1 + (size_t)idx * 32;
  for (int oc = 0; oc < 32; ++oc) {
    float s = c0[oc];
#pragma unroll
    for (int tap = 0; tap < 9; ++tap) s = fmaf(in[tap], K0[tap * 32 + oc], s);
    o[oc] = (_Float16)geluf(s);
  }
}

// ---------------------------------------------------------------------------
// conv2: 32->32 WMMA implicit GEMM; 8 waves/block, filters staged in LDS
// ---------------------------------------------------------------------------
__global__ void k_conv2_wmma(const _Float16* __restrict__ h1,
                             const _Float16* __restrict__ Wt2,
                             const float* __restrict__ c1,
                             _Float16* __restrict__ h2) {
  __shared__ _Float16 sWt[9 * 32 * 32];
  for (int i = threadIdx.x; i < (9 * 32 * 32 * 2) / 16; i += 256)
    ((uint4*)sWt)[i] = ((const uint4*)Wt2)[i];
  __syncthreads();

  const int lane = threadIdx.x & 31;
  const int id = blockIdx.x * 8 + (threadIdx.x >> 5);   // XT*128*BATCH tiles
  const int x0 = (id % XT) * 16;
  const int y = (id / XT) % 128;
  const int b = id / (XT * 128);
  const int xm = x0 + (lane & 15);
  v8f acc0 = {}, acc1 = {};
#pragma unroll
  for (int dy = -1; dy <= 1; ++dy) {
    int yy = y + dy;
    if (yy < 0 || yy >= 128) continue;
#pragma unroll
    for (int dx = -1; dx <= 1; ++dx) {
      int xs = xm + dx;
      v16h a = {};
      if (xs >= 0 && xs < NFRAMES)
        a = ldA16(h1 + (((size_t)b * 128 + yy) * NFRAMES + xs) * 32, lane);
      const _Float16* wb = sWt + (size_t)((dy + 1) * 3 + (dx + 1)) * 32 * 32;
      v16h b0 = ldB16(wb + (size_t)(lane & 15) * 32, lane);
      v16h b1 = ldB16(wb + (size_t)(16 + (lane & 15)) * 32, lane);
      acc0 = wmma_f16(a, b0, acc0);
      acc1 = wmma_f16(a, b1, acc1);
    }
  }
  const int n = lane & 15;
  const int mb = (lane & 16) ? 8 : 0;
#pragma unroll
  for (int r = 0; r < 8; ++r) {
    int xo = x0 + mb + r;
    if (xo < NFRAMES) {
      _Float16* o = h2 + (((size_t)b * 128 + y) * NFRAMES + xo) * 32;
      o[n]      = (_Float16)geluf(acc0[r] + c1[n]);
      o[16 + n] = (_Float16)geluf(acc1[r] + c1[16 + n]);
    }
  }
}

// ---------------------------------------------------------------------------
// conv3: 32->1 direct, GELU; stores dense-A layout (b*200+t) x 128 f16
// ---------------------------------------------------------------------------
__global__ void k_conv3(const _Float16* __restrict__ h2, const float* __restrict__ K2,
                        const float* __restrict__ c2, _Float16* __restrict__ dA) {
  int idx = blockIdx.x * blockDim.x + threadIdx.x;
  if (idx >= BATCH * 128 * NFRAMES) return;
  int xp = idx % NFRAMES, y = (idx / NFRAMES) % 128, b = idx / (128 * NFRAMES);
  float s = c2[0];
#pragma unroll
  for (int dy = 0; dy < 3; ++dy) {
    int yy = y + dy - 1;
    if (yy < 0 || yy >= 128) continue;
#pragma unroll
    for (int dx = 0; dx < 3; ++dx) {
      int xx = xp + dx - 1;
      if (xx < 0 || xx >= NFRAMES) continue;
      const _Float16* p = h2 + (((size_t)b * 128 + yy) * NFRAMES + xx) * 32;
      const float* wk = K2 + (size_t)(dy * 3 + dx) * 32 * 32;
      for (int ic = 0; ic < 32; ++ic) s = fmaf((float)p[ic], wk[ic * 32], s);
    }
  }
  dA[((size_t)b * NFRAMES + xp) * 128 + y] = (_Float16)geluf(s);
}

// ---------------------------------------------------------------------------
// dense + bias + sigmoid -> mask[3200][144]  (WMMA, K=128)
// Wdt (144x128 f16, 36KB) staged into LDS by the Tensor Data Mover.
// ---------------------------------------------------------------------------
__global__ void k_dense_mask(const _Float16* __restrict__ A,
                             const _Float16* __restrict__ Wdt,
                             const float* __restrict__ bd,
                             float* __restrict__ mask) {
  __shared__ _Float16 sW[NPAD * 128];
#if __has_builtin(__builtin_amdgcn_tensor_load_to_lds)
  typedef unsigned int u32x4 __attribute__((ext_vector_type(4)));
  typedef int i32x4 __attribute__((ext_vector_type(4)));
  typedef int i32x8 __attribute__((ext_vector_type(8)));
  if (threadIdx.x < 32) {
    const unsigned long long ga = (unsigned long long)(uintptr_t)Wdt;
    const unsigned int lds = (unsigned int)(uintptr_t)&sW[0];
    const unsigned int nelem = (NPAD * 128 * 2) / 8;  // 4608 8-byte elements
    u32x4 g0;
    g0[0] = 1u;                                           // count=1
    g0[1] = lds;                                          // lds_addr
    g0[2] = (unsigned int)(ga & 0xFFFFFFFFu);             // global_addr lo
    g0[3] = (unsigned int)((ga >> 32) & 0x01FFFFFFu) | (2u << 30);  // hi | type=2
    i32x8 g1;
    g1[0] = (int)(3u << 16);                              // data_size = 8B
    g1[1] = (int)((nelem & 0xFFFFu) << 16);               // tensor_dim0 lo
    g1[2] = (int)((nelem >> 16) | (1u << 16));            // dim0 hi | tensor_dim1=1
    g1[3] = (int)((nelem & 0xFFFFu) << 16);               // tile_dim0
    g1[4] = 1;                                            // tile_dim1=1
    g1[5] = (int)nelem;                                   // tensor_dim0_stride
    g1[6] = 0; g1[7] = 0;
    i32x4 z4 = {0, 0, 0, 0};
#if __has_include(<hip/amd_detail/amd_gfx1250_TDM.h>)
    i32x8 z8 = {0, 0, 0, 0, 0, 0, 0, 0};
    __builtin_amdgcn_tensor_load_to_lds(g0, g1, z4, z4, z8, 0);
#else
    __builtin_amdgcn_tensor_load_to_lds(g0, g1, z4, z4, 0);
#endif
#if __has_builtin(__builtin_amdgcn_s_wait_tensorcnt)
    __builtin_amdgcn_s_wait_tensorcnt(0);
#else
    asm volatile("s_wait_tensorcnt 0x0" ::: "memory");
#endif
  }
#else
  for (int i = threadIdx.x; i < (NPAD * 128 * 2) / 16; i += 256)
    ((uint4*)sW)[i] = ((const uint4*)Wdt)[i];
#endif
  __syncthreads();

  const int lane = threadIdx.x & 31;
  const int tid = blockIdx.x * 8 + (threadIdx.x >> 5);    // 1800 tiles exactly
  const int m0 = (tid % (NROWS / 16)) * 16;
  const int n0 = (tid / (NROWS / 16)) * 16;
  const _Float16* arow = A + (size_t)(m0 + (lane & 15)) * 128;
  const _Float16* brow = sW + (size_t)(n0 + (lane & 15)) * 128;
  v8f acc = {};
#pragma unroll
  for (int kk = 0; kk < 128; kk += 32)
    acc = wmma_f16(ldA16(arow + kk, lane), ldB16(brow + kk, lane), acc);
  const int n = n0 + (lane & 15);
  const float bias = (n < N_CH) ? bd[n] : 0.0f;
  const int mb = m0 + ((lane & 16) ? 8 : 0);
#pragma unroll
  for (int r = 0; r < 8; ++r)
    mask[(size_t)(mb + r) * NPAD + n] = sigmoidf(acc[r] + bias);
}

// ---------------------------------------------------------------------------
// masked spectrum -> f16 iSTFT A-matrix A2[3200][272]
// ---------------------------------------------------------------------------
__global__ void k_mask_apply(const float* __restrict__ spec,
                             const float* __restrict__ mask,
                             _Float16* __restrict__ A2) {
  int idx = blockIdx.x * blockDim.x + threadIdx.x;
  if (idx >= NROWS * 136) return;
  int row = idx / 136, k = idx % 136;
  _Float16 re = (_Float16)0.0f, im = (_Float16)0.0f;
  if (k < N_CH) {
    float m = mask[(size_t)row * NPAD + k];
    re = (_Float16)(spec[(size_t)row * 2 * NPAD + k] * m);
    im = (_Float16)(spec[(size_t)row * 2 * NPAD + NPAD + k] * m);
  }
  A2[(size_t)row * KSPEC + k] = re;
  A2[(size_t)row * KSPEC + 136 + k] = im;
}

// ---------------------------------------------------------------------------
// overlap-add + normalize
// ---------------------------------------------------------------------------
__global__ void k_zero(float* __restrict__ out, int n) {
  int i = blockIdx.x * blockDim.x + threadIdx.x;
  if (i < n) out[i] = 0.0f;
}
__global__ void k_ola(const float* __restrict__ yfr, float* __restrict__ out, int OT) {
  int idx = blockIdx.x * blockDim.x + threadIdx.x;
  if (idx >= NROWS * NFFT) return;
  int row = idx >> 8, n = idx & 255;
  int b = row / NFRAMES, t = row % NFRAMES;
  int s = t * HOP + n - NFFT / 2;
  if (s >= 0 && s < OT) atomicAdd(&out[(size_t)b * OT + s], yfr[idx]);
}
__global__ void k_norm(float* __restrict__ out, const float* __restrict__ denom,
                       int OT, int total) {
  int i = blockIdx.x * blockDim.x + threadIdx.x;
  if (i < total) out[i] = out[i] / denom[i % OT];
}

// ---------------------------------------------------------------------------
// host launcher
// ---------------------------------------------------------------------------
static inline char* carve(char*& p, size_t bytes) {
  char* r = p;
  p += (bytes + 255) & ~(size_t)255;
  return r;
}

extern "C" void kernel_launch(void* const* d_in, const int* in_sizes, int n_in,
                              void* d_out, int out_size, void* d_ws, size_t ws_size,
                              hipStream_t stream) {
  const float* x  = (const float*)d_in[0];
  const float* Bs = (const float*)d_in[1];
  const float* As = (const float*)d_in[2];
  const float* K0 = (const float*)d_in[3];
  const float* c0 = (const float*)d_in[4];
  const float* K1 = (const float*)d_in[5];
  const float* c1 = (const float*)d_in[6];
  const float* K2 = (const float*)d_in[7];
  const float* c2 = (const float*)d_in[8];
  const float* Wd = (const float*)d_in[9];
  const float* bd = (const float*)d_in[10];
  float* out = (float*)d_out;
  const int OT = out_size / BATCH;

  char* p = (char*)d_ws;
  _Float16* F      = (_Float16*)carve(p, (size_t)NROWS * NFFT * 2);
  _Float16* BtDFT  = (_Float16*)carve(p, (size_t)2 * NPAD * NFFT * 2);
  float*    spec   = (float*)   carve(p, (size_t)NROWS * 2 * NPAD * 4);
  float*    aud    = (float*)   carve(p, (size_t)BATCH * 128 * NFRAMES * 4);
  _Float16* h1     = (_Float16*)carve(p, (size_t)BATCH * 128 * NFRAMES * 32 * 2);
  _Float16* h2     = (_Float16*)carve(p, (size_t)BATCH * 128 * NFRAMES * 32 * 2);
  _Float16* Wt2    = (_Float16*)carve(p, (size_t)9 * 32 * 32 * 2);
  _Float16* dA     = (_Float16*)carve(p, (size_t)NROWS * 128 * 2);
  _Float16* Wdt    = (_Float16*)carve(p, (size_t)NPAD * 128 * 2);
  float*    mask   = (float*)   carve(p, (size_t)NROWS * NPAD * 4);
  _Float16* A2     = (_Float16*)carve(p, (size_t)NROWS * KSPEC * 2);
  _Float16* Bt2    = (_Float16*)carve(p, (size_t)NFFT * KSPEC * 2);
  float*    yfr    = (float*)   carve(p, (size_t)NROWS * NFFT * 4);
  float*    denom  = (float*)   carve(p, (size_t)(OT > 0 ? OT : 1) * 4);

  // tables
  k_build_dft  <<<dim3(2 * NPAD), dim3(NFFT), 0, stream>>>(BtDFT);
  k_build_idft <<<dim3(NFFT), dim3(KSPEC), 0, stream>>>(Bt2);
  {
    int tot = 9 * 32 * 32 + NPAD * 128;
    k_build_weights<<<dim3((tot + 255) / 256), dim3(256), 0, stream>>>(K1, Wd, Wt2, Wdt);
  }
  k_build_denom<<<dim3((OT + 255) / 256), dim3(256), 0, stream>>>(denom, OT);

  // STFT path: frames -> WMMA GEMM (M=3200, K=256, N=288) : 3600 tiles / 8
  k_frames<<<dim3((NROWS * NFFT + 255) / 256), dim3(256), 0, stream>>>(x, F);
  k_gemm_f16<<<dim3((NROWS / 16) * (2 * NPAD / 16) / 8), dim3(256), 0, stream>>>(
      F, BtDFT, spec, NFFT, 2 * NPAD, NROWS / 16, 2 * NPAD / 16);

  // cochlear model
  k_iir_aud<<<dim3(BATCH), dim3(128), 0, stream>>>(x, Bs, As, aud);

  // conv stack
  k_conv1<<<dim3((BATCH * 128 * NFRAMES + 255) / 256), dim3(256), 0, stream>>>(
      aud, K0, c0, h1);
  k_conv2_wmma<<<dim3(XT * 128 * BATCH / 8), dim3(256), 0, stream>>>(h1, Wt2, c1, h2);
  k_conv3<<<dim3((BATCH * 128 * NFRAMES + 255) / 256), dim3(256), 0, stream>>>(
      h2, K2, c2, dA);

  // dense + sigmoid mask (M=3200, K=128, N=144) : 1800 tiles / 8
  k_dense_mask<<<dim3((NROWS / 16) * (NPAD / 16) / 8), dim3(256), 0, stream>>>(
      dA, Wdt, bd, mask);

  // iSTFT: mask*spec -> f16, GEMM (M=3200, K=272, N=256) : 3200 tiles / 8
  k_mask_apply<<<dim3((NROWS * 136 + 255) / 256), dim3(256), 0, stream>>>(spec, mask, A2);
  k_gemm_f16<<<dim3((NROWS / 16) * (NFFT / 16) / 8), dim3(256), 0, stream>>>(
      A2, Bt2, yfr, KSPEC, NFFT, NROWS / 16, NFFT / 16);
  k_zero<<<dim3((out_size + 255) / 256), dim3(256), 0, stream>>>(out, out_size);
  k_ola <<<dim3((NROWS * NFFT + 255) / 256), dim3(256), 0, stream>>>(yfr, out, OT);
  k_norm<<<dim3((out_size + 255) / 256), dim3(256), 0, stream>>>(out, denom, OT, out_size);
}
